// MatchingLayer_63531156242859
// MI455X (gfx1250) — matrix operand order repaired and checked by reading, not compiled, not run on hardware
//
#include <hip/hip_runtime.h>
#include <hip/hip_bf16.h>

typedef __attribute__((ext_vector_type(16))) _Float16 v16h;
typedef __attribute__((ext_vector_type(8)))  _Float16 v8h;
typedef __attribute__((ext_vector_type(8)))  float    v8f;

#define N_PIX 9216   // 96*96 pixels
#define C_DIM 256    // channels
#define TOPK  20

// ---------------------------------------------------------------------------
// Kernel 1: L2-normalize per-pixel feature vectors and transpose to row-major
// f16.  Input is [C][N] (channel-major); output is [N][C] row-major f16 so
// WMMA fragment loads in kernel 2 are contiguous 16/32-byte loads.
// ---------------------------------------------------------------------------
__global__ __launch_bounds__(256) void norm_f16_kernel(
    const float* __restrict__ x, _Float16* __restrict__ out) {
  int n = blockIdx.x * blockDim.x + threadIdx.x;
  if (n >= N_PIX) return;
  float ss = 0.0f;
#pragma unroll 8
  for (int c = 0; c < C_DIM; ++c) {
    float v = x[c * N_PIX + n];   // coalesced across threads
    ss += v * v;
  }
  float inv = 1.0f / fmaxf(sqrtf(ss), 1e-12f);
#pragma unroll 8
  for (int c = 0; c < C_DIM; ++c) {
    out[n * C_DIM + c] = (_Float16)(x[c * N_PIX + n] * inv);
  }
}

// ---------------------------------------------------------------------------
// Kernel 2: fused cosine-similarity GEMM (WMMA f16 -> f32 accumulate) + per-
// row top-K, software-pipelined: B fragments for tile t+1 are loaded before
// the WMMAs + top-K insertion of tile t, so the insertion phase hides the
// full B-load latency.  One wave32 owns 16 s-rows; A fragments (K=256) stay
// hoisted in VGPRs for the whole sweep.
// ---------------------------------------------------------------------------
__global__ __launch_bounds__(256) void sim_topk_kernel(
    const _Float16* __restrict__ Sh,   // [N_PIX][C_DIM] normalized s rows
    const _Float16* __restrict__ Qh,   // [N_PIX][C_DIM] normalized q rows
    const unsigned char* __restrict__ mask,  // [N_PIX] fg indicator (q pixels)
    float* __restrict__ out) {         // [2*N_PIX]: fg map then bg map
  __shared__ float tile[8][16][16];
  __shared__ float topf[8][16][TOPK];
  __shared__ float topb[8][16][TOPK];
  __shared__ unsigned char lmask[N_PIX];   // 9 KB, shared by all 8 waves

  const int w    = threadIdx.x >> 5;   // wave within block
  const int lane = threadIdx.x & 31;
  const int r0   = (blockIdx.x * 8 + w) * 16;  // first s-row of this wave

  // Cooperative whole-mask preload (word-wise: 2304 b32 loads / block).
  {
    const unsigned int* m32 = (const unsigned int*)mask;
    unsigned int* lm32 = (unsigned int*)lmask;
    for (int i = threadIdx.x; i < N_PIX / 4; i += 256) lm32[i] = m32[i];
  }

  // Lane L owns (row L&15, class L>>4): 0 = fg list, 1 = bg list.
  const int r   = lane & 15;
  const int cls = lane >> 4;
  float* lst = cls ? &topb[w][r][0] : &topf[w][r][0];
  const float NEG_INF = -__builtin_huge_valf();
#pragma unroll
  for (int k = 0; k < TOPK; ++k) lst[k] = NEG_INF;
  __syncthreads();

  // Per-lane WMMA fragment addressing (16-bit A 16x32 / B 32x16 layouts):
  //   lanes 0-15 : A elems 0..15 = K c0+0..15
  //   lanes16-31 : A elems 0..15 = K c0+16..31
  //   lanes 0-15 : B elems 0..15 = K c0+0..15 ; lanes16-31: K c0+16..31
  const int hi = cls * 8;                    // 0 or 8
  const _Float16* arow = Sh + (size_t)(r0 + r) * C_DIM;

  // Hoist all 8 A fragments (this wave's 16 s-rows, K=0..255) into VGPRs.
  v16h afrag[8];
#pragma unroll
  for (int kt = 0; kt < 8; ++kt) {
    const int c0 = kt * 32;
    const v8h a0 = *(const v8h*)(arow + c0 + 2 * hi);
    const v8h a1 = *(const v8h*)(arow + c0 + 2 * hi + 8);
#pragma unroll
    for (int i = 0; i < 8; ++i) { afrag[kt][i] = a0[i]; afrag[kt][8 + i] = a1[i]; }
  }

  // Lane's base into a B column tile starting at q-pixel m.
  const _Float16* qlane = Qh + (size_t)r * C_DIM + 2 * hi;

  auto load_b = [&](v16h (&bf)[8], int m) {
    const _Float16* bcol = qlane + (size_t)m * C_DIM;
#pragma unroll
    for (int kt = 0; kt < 8; ++kt)
      bf[kt] = *(const v16h*)(bcol + kt * 32);
  };

  // 8 WMMAs + LDS spill + class-split sorted top-K insertion for one tile.
  auto process = [&](const v16h (&bf)[8], int m0) {
    v8f acc = {};
#pragma unroll
    for (int kt = 0; kt < 8; ++kt)
      acc = __builtin_amdgcn_wmma_f32_16x16x32_f16(
          false, afrag[kt], false, bf[kt], (short)0, acc, false, false);
    // C layout: VGPR i -> (M = cls*8 + i, N = lane&15)
#pragma unroll
    for (int i = 0; i < 8; ++i) tile[w][cls * 8 + i][r] = acc[i];
    __syncthreads();
    const unsigned char* mt = &lmask[m0];
    for (int n = 0; n < 16; ++n) {
      const float v = tile[w][r][n];
      if ((mt[n] != 0) == (cls == 0)) {        // class test, uniform per 16-lane half
        if (v > lst[0]) {
          int i = 0;
          while (i < TOPK - 1 && lst[i + 1] < v) { lst[i] = lst[i + 1]; ++i; }
          lst[i] = v;
        }
      }
    }
    __syncthreads();
  };

  // Software pipeline, unrolled by 2 with ping-pong B buffers (576 tiles).
  v16h b0[8], b1[8];
  load_b(b0, 0);
  for (int m0 = 0; m0 < N_PIX; m0 += 32) {
    const int mNext = (m0 + 32 < N_PIX) ? m0 + 32 : 0;  // tail: dummy reload
    if (m0 + 64 < N_PIX)
      __builtin_prefetch(qlane + (size_t)(m0 + 64) * C_DIM, 0, 1);
    load_b(b1, m0 + 16);     // in flight across process(b0)
    process(b0, m0);
    load_b(b0, mNext);       // in flight across process(b1)
    process(b1, m0 + 16);
  }

  // Mean of each lane's 20-element list; fg -> out[0..], bg -> out[N_PIX..].
  {
    float s = 0.0f;
#pragma unroll
    for (int k = 0; k < TOPK; ++k) s += lst[k];
    out[cls * N_PIX + r0 + r] = s * (1.0f / TOPK);
  }
}

// ---------------------------------------------------------------------------
// Launch: inputs in setup_inputs() order: mask (bool), q_feat, s_feat (f32).
// d_out = [fg_score(9216) | bg_score(9216)] f32.  d_ws holds Qh|Sh f16.
// ---------------------------------------------------------------------------
extern "C" void kernel_launch(void* const* d_in, const int* in_sizes, int n_in,
                              void* d_out, int out_size, void* d_ws, size_t ws_size,
                              hipStream_t stream) {
  (void)in_sizes; (void)n_in; (void)out_size; (void)ws_size;
  const unsigned char* mask = (const unsigned char*)d_in[0];
  const float* q_feat = (const float*)d_in[1];
  const float* s_feat = (const float*)d_in[2];
  float* out = (float*)d_out;

  _Float16* Qh = (_Float16*)d_ws;                       // 9216*256 f16
  _Float16* Sh = Qh + (size_t)N_PIX * C_DIM;            // 9216*256 f16

  norm_f16_kernel<<<N_PIX / 256, 256, 0, stream>>>(q_feat, Qh);
  norm_f16_kernel<<<N_PIX / 256, 256, 0, stream>>>(s_feat, Sh);

  // 576 row-blocks of 16 s-rows, 8 waves (of 32) per 256-thread block.
  sim_topk_kernel<<<(N_PIX / 16) / 8, 256, 0, stream>>>(Sh, Qh, mask, out);
}
